// TGNMemory_3075196584344
// MI455X (gfx1250) — compile-verified, matrix operand order checked by reading, not run on hardware
//
#include <hip/hip_runtime.h>
#include <hip/hip_bf16.h>

// ---------------------------------------------------------------------------
// TGN memory update for MI455X (gfx1250, wave32, WMMA f16 16x16x32).
// D = 128, RAW = 128, T = 128, MSG = 512, 3D = 384.
//
// Pipeline: memset scratch -> pack weights (f16 B-fragments) ->
//           event scatter (pk_add_f16 atomics) -> GRU (WMMA, 32 nodes/block)
// ---------------------------------------------------------------------------

typedef __attribute__((ext_vector_type(16))) _Float16 v16h;
typedef __attribute__((ext_vector_type(8)))  float    v8f;

#define DD     128
#define MSGW   512
#define SUMW   384                 // f16 scatter width (nbr-mem | raw | time-enc)
#define SUMW_U 192                 // row length in u32 (2 f16 each)
#define AGGR_STRIDE_U 260          // 256 u32 (512 f16) + 4 pad dwords -> conflict-free
#define MEMH_STRIDE_U 68           // 64 u32 (128 f16)  + 4 pad dwords
#define MT     32                  // nodes per GRU block (2 row-tiles)

union FragU { unsigned u[8]; uint4 q[2]; v16h h; };
union H2U   { unsigned u; _Float16 h[2]; };

static __device__ inline v8f wmma16(v16h a, v16h b, v8f c) {
  return __builtin_amdgcn_wmma_f32_16x16x32_f16(false, a, false, b, (short)0, c,
                                                false, false);
}

// B fragment, pre-swizzled in global scratch: [frag][lane] = 32 contiguous bytes.
static __device__ inline v16h load_bfrag(const uint4* base, int frag, int lane) {
  FragU f;
  const uint4* p = base + (size_t)frag * 64 + lane * 2;
  f.q[0] = p[0];
  f.q[1] = p[1];
  return f.h;
}

// A fragment from LDS (f16 rows, padded stride). 16-bit A 16x32 layout:
// row m = lane&15, koff = (lane>>4)*8; vgpr p: K = kbase + (p&3)*2 + koff (+16 if p>=4)
static __device__ inline v16h load_afrag(const unsigned* rows, int strideU,
                                         int m, int koff, int kbase) {
  FragU f;
  const unsigned* r = rows + m * strideU;
  int i0 = (kbase + koff) >> 1;        // u32 index, multiple of 4 -> ds_load_b128
  f.q[0] = *(const uint4*)(r + i0);
  f.q[1] = *(const uint4*)(r + i0 + 8);
  return f.h;
}

static __device__ inline unsigned pack2h(float a, float b) {
  H2U c;
  c.h[0] = (_Float16)a; c.h[1] = (_Float16)b;
  return c.u;
}

// Native CDNA5 packed-f16 atomic add (no return -> STOREcnt, no CAS loop).
static __device__ inline void gatomic_pk_f16(unsigned* p, unsigned v) {
  asm volatile("global_atomic_pk_add_f16 %0, %1, off" :: "v"(p), "v"(v) : "memory");
}

// ---------------------------------------------------------------------------
// Kernel 1: pre-pack w_ih (384x512) and w_hh (384x128) into f16 B-fragments.
// gx frags: 24 jtiles * 16 ksteps = 384;  gh frags: 24 * 4 = 96. 32 thr/frag.
// B[k][n] = W[jtile*16 + n][k];  lane: n = lane&15, koff = (lane>>4)*8.
// ---------------------------------------------------------------------------
__global__ void pack_weights_kernel(const float* __restrict__ w_ih,
                                    const float* __restrict__ w_hh,
                                    unsigned* __restrict__ packIH,
                                    unsigned* __restrict__ packHH) {
  int f    = blockIdx.x;
  int lane = threadIdx.x;
  int n    = lane & 15;
  int koff = (lane >> 4) * 8;
  if (f < 384) {
    int jtile = f >> 4, ks = f & 15;
    const float* row = w_ih + (size_t)(jtile * 16 + n) * MSGW;
    unsigned* outp = packIH + (size_t)f * 256 + lane * 8;
#pragma unroll
    for (int p = 0; p < 8; ++p) {
      int k0 = ks * 32 + (p & 3) * 2 + koff + ((p & 4) ? 16 : 0);
      outp[p] = pack2h(row[k0], row[k0 + 1]);
    }
  } else {
    int f2 = f - 384;
    int jtile = f2 >> 2, ks = f2 & 3;
    const float* row = w_hh + (size_t)(jtile * 16 + n) * DD;
    unsigned* outp = packHH + (size_t)f2 * 256 + lane * 8;
#pragma unroll
    for (int p = 0; p < 8; ++p) {
      int k0 = ks * 32 + (p & 3) * 2 + koff + ((p & 4) ? 16 : 0);
      outp[p] = pack2h(row[k0], row[k0 + 1]);
    }
  }
}

// ---------------------------------------------------------------------------
// Kernel 2: per-event scatter. 4 events per 256-thread block, 64 thr/event,
// each thread owns a dim pair -> one pk_add_f16 atomic per 2 dims.
// sumsH[n][0:128)=sum neighbor memory, [128:256)=sum raw, [256:384)=sum enc.
// (the "self memory" block of the message reduces to cnt*mem -> no atomics)
// ---------------------------------------------------------------------------
__global__ void scatter_kernel(const float* __restrict__ memory,
                               const int*   __restrict__ last_update,
                               const int*   __restrict__ src,
                               const int*   __restrict__ dst,
                               const int*   __restrict__ tvec,
                               const float* __restrict__ raw,
                               const float* __restrict__ time_w,
                               const float* __restrict__ time_b,
                               unsigned* __restrict__ sumsH,
                               int*      __restrict__ cnt,
                               int*      __restrict__ lum,
                               int E) {
  int tid = threadIdx.x;
  int e   = blockIdx.x * 4 + (tid >> 6);
  int d2  = tid & 63;                        // dim pair index (dims 2*d2, 2*d2+1)
  if (e >= E) return;

  int s  = src[e];
  int dd = dst[e];
  int tt = tvec[e];

  float2 ms = ((const float2*)memory)[(size_t)s  * 64 + d2];
  float2 md = ((const float2*)memory)[(size_t)dd * 64 + d2];
  float2 rw = ((const float2*)raw)[(size_t)e * 64 + d2];
  float2 tw = ((const float2*)time_w)[d2];
  float2 tb = ((const float2*)time_b)[d2];
  float dts = (float)(tt - last_update[s]);
  float dtd = (float)(tt - last_update[dd]);
  unsigned encS = pack2h(__cosf(dts * tw.x + tb.x), __cosf(dts * tw.y + tb.y));
  unsigned encD = pack2h(__cosf(dtd * tw.x + tb.x), __cosf(dtd * tw.y + tb.y));
  unsigned msP  = pack2h(ms.x, ms.y);
  unsigned mdP  = pack2h(md.x, md.y);
  unsigned rwP  = pack2h(rw.x, rw.y);

  unsigned* rs = sumsH + (size_t)s  * SUMW_U + d2;
  unsigned* rd = sumsH + (size_t)dd * SUMW_U + d2;
  gatomic_pk_f16(rs,       mdP);    // neighbor memory into src store
  gatomic_pk_f16(rs + 64,  rwP);    // raw msg
  gatomic_pk_f16(rs + 128, encS);   // time encoding
  gatomic_pk_f16(rd,       msP);    // neighbor memory into dst store
  gatomic_pk_f16(rd + 64,  rwP);
  gatomic_pk_f16(rd + 128, encD);

  if (d2 == 0) {
    atomicAdd(cnt + s, 1);
    atomicAdd(cnt + dd, 1);
    atomicMax(lum + s, tt);
    atomicMax(lum + dd, tt);
  }
}

// ---------------------------------------------------------------------------
// Kernel 3: GRU cell via WMMA. 1 block = 32 nodes (2 row-tiles), 8 waves.
// Wave jc owns output columns [jc*16, jc*16+16) for all 3 gates and BOTH
// row-tiles -> every B fragment load feeds 2 WMMAs (halves L2 B-traffic).
// ---------------------------------------------------------------------------
__global__ __launch_bounds__(256)
void gru_kernel(const float* __restrict__ memory,
                const unsigned* __restrict__ sumsH,
                const int*   __restrict__ cnt,
                const int*   __restrict__ lum,
                const uint4* __restrict__ packIH,
                const uint4* __restrict__ packHH,
                const float* __restrict__ b_ih,
                const float* __restrict__ b_hh,
                float* __restrict__ outMem,
                float* __restrict__ outLU) {
  __shared__ __align__(16) unsigned aggrU[MT * AGGR_STRIDE_U]; // 32 x 512 f16 (+pad)
  __shared__ __align__(16) unsigned memU [MT * MEMH_STRIDE_U]; // 32 x 128 f16 (+pad)
  __shared__ int scnt[MT];

  int tid = threadIdx.x;
  int nodeBase = blockIdx.x * MT;

  if (tid < MT) scnt[tid] = cnt[nodeBase + tid];
  __syncthreads();

  // Stage A-matrix for gx: cols [0,128)  = self memory (0 if cnt==0),
  //                        cols [128,512) = f16 sums * 1/max(cnt,1)
#pragma unroll 4
  for (int i = 0; i < MT * 256 / 256; ++i) {   // 32 iters, 8192 u32 total
    int q = tid + i * 256;
    int m  = q >> 8;
    int kk = q & 255;                 // u32 within row
    int k  = kk * 2;
    int node = nodeBase + m;
    int c = scnt[m];
    float a0, a1;
    if (k < DD) {
      float sc = (c > 0) ? 1.0f : 0.0f;
      const float2 mv = ((const float2*)memory)[(size_t)node * 64 + kk];
      a0 = mv.x * sc;
      a1 = mv.y * sc;
    } else {
      float inv = 1.0f / (float)(c > 0 ? c : 1);
      H2U sv; sv.u = sumsH[(size_t)node * SUMW_U + ((k - DD) >> 1)];
      a0 = (float)sv.h[0] * inv;
      a1 = (float)sv.h[1] * inv;
    }
    aggrU[m * AGGR_STRIDE_U + kk] = pack2h(a0, a1);
  }
  // Stage A-matrix for gh: memory in f16
#pragma unroll 4
  for (int i = 0; i < 8; ++i) {                // 2048 u32 total
    int q = tid + i * 256;
    int m  = q >> 6;
    int kk = q & 63;
    const float2 mv = ((const float2*)memory)[(size_t)(nodeBase + m) * 64 + kk];
    memU[m * MEMH_STRIDE_U + kk] = pack2h(mv.x, mv.y);
  }
  __syncthreads();

  int wave = tid >> 5;                // jc = column group 0..7
  int lane = tid & 31;
  int mrow = lane & 15;
  int koff = (lane >> 4) * 8;
  int jc   = wave;

  v8f z8 = {0,0,0,0,0,0,0,0};
  v8f xr0 = z8, xz0 = z8, xn0 = z8, hr0 = z8, hz0 = z8, hn0 = z8;
  v8f xr1 = z8, xz1 = z8, xn1 = z8, hr1 = z8, hz1 = z8, hn1 = z8;

  // gx = aggr @ w_ih^T   (K = 512)
#pragma unroll
  for (int ks = 0; ks < 16; ++ks) {
    v16h a0 = load_afrag(aggrU, AGGR_STRIDE_U, mrow,      koff, ks * 32);
    v16h a1 = load_afrag(aggrU, AGGR_STRIDE_U, mrow + 16, koff, ks * 32);
    v16h br = load_bfrag(packIH, (jc)      * 16 + ks, lane);
    v16h bz = load_bfrag(packIH, (8  + jc) * 16 + ks, lane);
    v16h bn = load_bfrag(packIH, (16 + jc) * 16 + ks, lane);
    xr0 = wmma16(a0, br, xr0);  xr1 = wmma16(a1, br, xr1);
    xz0 = wmma16(a0, bz, xz0);  xz1 = wmma16(a1, bz, xz1);
    xn0 = wmma16(a0, bn, xn0);  xn1 = wmma16(a1, bn, xn1);
  }
  // gh = memory @ w_hh^T (K = 128)
#pragma unroll
  for (int ks = 0; ks < 4; ++ks) {
    v16h a0 = load_afrag(memU, MEMH_STRIDE_U, mrow,      koff, ks * 32);
    v16h a1 = load_afrag(memU, MEMH_STRIDE_U, mrow + 16, koff, ks * 32);
    v16h br = load_bfrag(packHH, (jc)      * 4 + ks, lane);
    v16h bz = load_bfrag(packHH, (8  + jc) * 4 + ks, lane);
    v16h bn = load_bfrag(packHH, (16 + jc) * 4 + ks, lane);
    hr0 = wmma16(a0, br, hr0);  hr1 = wmma16(a1, br, hr1);
    hz0 = wmma16(a0, bz, hz0);  hz1 = wmma16(a1, bz, hz1);
    hn0 = wmma16(a0, bn, hn0);  hn1 = wmma16(a1, bn, hn1);
  }

  // Epilogue: biases + gates. C/D layout: lane -> col = lane&15,
  // vgpr r -> row m = r + (lane>>4)*8 within the 16-row tile.
  int col = jc * 16 + (lane & 15);
  float bir = b_ih[col], biz = b_ih[DD + col], bin = b_ih[2 * DD + col];
  float bhr = b_hh[col], bhz = b_hh[DD + col], bhn = b_hh[2 * DD + col];

#pragma unroll
  for (int rt = 0; rt < 2; ++rt) {
    const v8f& xr = rt ? xr1 : xr0;  const v8f& hr = rt ? hr1 : hr0;
    const v8f& xz = rt ? xz1 : xz0;  const v8f& hz = rt ? hz1 : hz0;
    const v8f& xn = rt ? xn1 : xn0;  const v8f& hn = rt ? hn1 : hn0;
#pragma unroll
    for (int r = 0; r < 8; ++r) {
      int node = nodeBase + rt * 16 + r + (lane >> 4) * 8;
      float vr = xr[r] + bir + hr[r] + bhr;
      float vz = xz[r] + biz + hz[r] + bhz;
      float rg = 1.0f / (1.0f + __expf(-vr));
      float zg = 1.0f / (1.0f + __expf(-vz));
      float ng = tanhf(xn[r] + bin + rg * (hn[r] + bhn));
      float h  = memory[(size_t)node * DD + col];
      outMem[(size_t)node * DD + col] = (1.0f - zg) * ng + zg * h;
    }
  }

  if (tid < MT) {
    int v = lum[nodeBase + tid];
    outLU[nodeBase + tid] = (float)(v > 0 ? v : 0);
  }
}

// ---------------------------------------------------------------------------
extern "C" void kernel_launch(void* const* d_in, const int* in_sizes, int n_in,
                              void* d_out, int out_size, void* d_ws, size_t ws_size,
                              hipStream_t stream) {
  const float* memory      = (const float*)d_in[0];
  const int*   last_update = (const int*)  d_in[1];
  const int*   src         = (const int*)  d_in[2];
  const int*   dst         = (const int*)  d_in[3];
  const int*   tvec        = (const int*)  d_in[4];
  const float* raw         = (const float*)d_in[5];
  const float* time_w      = (const float*)d_in[6];
  const float* time_b      = (const float*)d_in[7];
  const float* w_ih        = (const float*)d_in[8];
  const float* w_hh        = (const float*)d_in[9];
  const float* b_ih        = (const float*)d_in[10];
  const float* b_hh        = (const float*)d_in[11];

  const int N = in_sizes[0] / DD;
  const int E = in_sizes[2];

  // Workspace layout (256B aligned sections)
  char* ws = (char*)d_ws;
  size_t off = 0;
  unsigned* sumsH = (unsigned*)(ws + off);               // N x 384 f16
  off += ((size_t)N * SUMW * sizeof(_Float16) + 255) & ~(size_t)255;
  int* cnt = (int*)(ws + off);
  off += ((size_t)N * sizeof(int) + 255) & ~(size_t)255;
  int* lum = (int*)(ws + off);
  off += ((size_t)N * sizeof(int) + 255) & ~(size_t)255;
  size_t zeroBytes = off;
  unsigned* packIH = (unsigned*)(ws + off);
  off += (size_t)384 * 1024;               // 384 frags x 1KB
  unsigned* packHH = (unsigned*)(ws + off);
  off += (size_t)96 * 1024;                // 96 frags x 1KB

  float* outMem = (float*)d_out;
  float* outLU  = outMem + (size_t)N * DD;

  hipMemsetAsync(ws, 0, zeroBytes, stream);

  pack_weights_kernel<<<480, 32, 0, stream>>>(w_ih, w_hh, packIH, packHH);

  scatter_kernel<<<(E + 3) / 4, 256, 0, stream>>>(
      memory, last_update, src, dst, tvec, raw, time_w, time_b,
      sumsH, cnt, lum, E);

  gru_kernel<<<N / MT, 256, 0, stream>>>(
      memory, sumsH, cnt, lum,
      (const uint4*)packIH, (const uint4*)packHH, b_ih, b_hh,
      outMem, outLU);
}
